// WaterMarkModule_87531433492737
// MI455X (gfx1250) — matrix operand order repaired and checked by reading, not compiled, not run on hardware
//
#include <hip/hip_runtime.h>
#include <math.h>

#define BSZ   2
#define NN    384
#define FEAT  64
#define CP    128
#define K1DIM 192
#define DEG   32
#define NEDGE (BSZ*NN*DEG)
#define CH    32
#define HH    384
#define WW    3
#define EPSV  1e-5f
#define PI_F  3.14159265358979f

typedef _Float16 f16;
typedef __attribute__((ext_vector_type(16))) _Float16 v16h;
typedef __attribute__((ext_vector_type(8)))  _Float16 v8h;
typedef __attribute__((ext_vector_type(8)))  float    v8f;

// ---------------------------------------------------------------- utilities
__global__ void zero_f32_kernel(float* p, int n) {
  int i = blockIdx.x * blockDim.x + threadIdx.x;
  if (i < n) p[i] = 0.0f;
}

// ------------------------------------------------- segment-sum of radial
__global__ void agg_kernel(const float* __restrict__ pos,
                           const int* __restrict__ eidx,
                           float* __restrict__ agg) {
  int e = blockIdx.x * blockDim.x + threadIdx.x;
  if (e >= NEDGE) return;
  int r = eidx[e];
  int c = eidx[NEDGE + e];
  float dx = pos[r*3+0] - pos[c*3+0];
  float dy = pos[r*3+1] - pos[c*3+1];
  float dz = pos[r*3+2] - pos[c*3+2];
  atomicAdd(&agg[r], dx*dx + dy*dy + dz*dz);
}

// ------------------------------------------------- per-node embeddings -> pi
__global__ void node_kernel(const float* __restrict__ charges,
                            const float* __restrict__ atom,
                            const float* __restrict__ w_aa, const float* __restrict__ b_aa,
                            const float* __restrict__ w_node, const float* __restrict__ b_node,
                            const float* __restrict__ w_sp, const float* __restrict__ b_sp,
                            const float* __restrict__ agg,
                            float* __restrict__ node_embed,
                            float* __restrict__ pi_out) {
  __shared__ float cat[256];
  __shared__ float ne[4];
  int node = blockIdx.x;          // 0 .. BSZ*NN-1
  int n    = node % NN;
  int t    = threadIdx.x;         // 128 threads
  float chg = charges[node];
  // positional embedding * charge  (128-dim)
  float emb;
  if (t < 64) emb = sinf((float)n * PI_F / powf(2056.0f, 2.0f * (float)t / 128.0f));
  else        emb = cosf((float)n * PI_F / powf(2056.0f, 2.0f * (float)(t-64) / 128.0f));
  cat[t] = emb * chg;
  // atom-type embedding (5 -> 128)
  float a = b_aa[t];
  for (int q = 0; q < 5; ++q) a += atom[node*5+q] * w_aa[q*128 + t];
  cat[128 + t] = a;
  __syncthreads();
  if (t < 3) {
    float s = b_node[t];
    for (int c = 0; c < 256; ++c) s += cat[c] * w_node[c*3 + t];
    ne[t] = s;
    node_embed[node*3 + t] = s;
  }
  if (t == 3) ne[3] = agg[node];
  __syncthreads();
  if (t < 64) {
    float s = b_sp[t];
    for (int d = 0; d < 4; ++d) s += ne[d] * w_sp[d*64 + t];
    pi_out[node*64 + t] = s;
  }
}

// ------------------------------------------------- relative-position table
__global__ void relfeat_kernel(const float* __restrict__ w_rp,
                               const float* __restrict__ b_rp,
                               float* __restrict__ relf) {
  __shared__ float emb[64];
  int r = blockIdx.x;             // 0..767 ; rel = r-383
  float rel = (float)(r - 383);
  int t = threadIdx.x;            // 64 threads
  if (t < 32) emb[t] = sinf(rel * PI_F / powf(2056.0f, 2.0f * (float)t / 64.0f));
  else        emb[t] = cosf(rel * PI_F / powf(2056.0f, 2.0f * (float)(t-32) / 64.0f));
  __syncthreads();
  float s = b_rp[t];
  for (int k = 0; k < 64; ++k) s += emb[k] * w_rp[k*64 + t];
  relf[r*64 + t] = s;
}

// ------------------------------------------------- pack weights into WMMA B layout
// B operand (16-bit, 32x16), in this code's contiguous k order: lane l holds col n=l&15, K = (l>>4)*16 + e, e=0..15
// packed[((s*8+t)*32 + l)*16 + e] = W[(s*32 + (l>>4)*16 + e)*128 + t*16 + (l&15)]
__global__ void pack_kernel(const float* __restrict__ w1,
                            const float* __restrict__ w2,
                            const float* __restrict__ w3,
                            f16* __restrict__ p1, f16* __restrict__ p2, f16* __restrict__ p3) {
  int gid = blockIdx.x * blockDim.x + threadIdx.x;  // 57344 total
  const float* src; f16* dst; int p;
  if (gid < 24576)       { src = w1; dst = p1; p = gid; }
  else if (gid < 40960)  { src = w2; dst = p2; p = gid - 24576; }
  else if (gid < 57344)  { src = w3; dst = p3; p = gid - 40960; }
  else return;
  int e   = p & 15;
  int l   = (p >> 4) & 31;
  int blk = p >> 9;
  int t   = blk & 7;
  int s   = blk >> 3;
  int k   = s*32 + ((l >> 4) * 16) + e;
  int col = t*16 + (l & 15);
  dst[p] = (f16)src[k*128 + col];
}

// ------------------------------------------------- WMMA MLP layer (per wave)
// A (16-bit, 16x32): lane l -> row m=l&15; halfs e=0..15 -> K=s*32+(l>>4)*16+e (same order as B)
// C/D f32: VGPR v, lane l -> row v+(l>>4)*8, col l&15
__device__ __forceinline__ void wmma_layer(f16* stg, const f16* __restrict__ pw,
                                           const float* __restrict__ bias,
                                           int nk, bool do_relu, int lane) {
  int m  = lane & 15;
  int hb = lane >> 4;
  v8f acc[8];
  for (int t = 0; t < 8; ++t)
    for (int e = 0; e < 8; ++e) acc[t][e] = 0.0f;
  for (int s = 0; s < nk; ++s) {
    const f16* ap = stg + m*K1DIM + s*32 + hb*16;
    v8h alo = *(const v8h*)ap;
    v8h ahi = *(const v8h*)(ap + 8);
    v16h a;
    for (int e = 0; e < 8; ++e) { a[e] = alo[e]; a[e+8] = ahi[e]; }
    for (int t = 0; t < 8; ++t) {
      v16h bm = *(const v16h*)(pw + (((s << 3) + t) * 32 + lane) * 16);
      acc[t] = __builtin_amdgcn_wmma_f32_16x16x32_f16(
          false, a, false, bm, (short)0, acc[t], false, false);
    }
  }
  asm volatile("s_wait_dscnt 0" ::: "memory");
  for (int t = 0; t < 8; ++t) {
    int col = (t << 4) + m;
    float bv = bias[col];
    for (int v = 0; v < 8; ++v) {
      float x = acc[t][v] + bv;
      if (do_relu && x < 0.0f) x = 0.0f;
      stg[(v + hb*8)*K1DIM + col] = (f16)x;
    }
  }
}

// ------------------------------------------------- edge MLP (the hot GEMM)
__global__ __launch_bounds__(256)
void edge_mlp_kernel(const float* __restrict__ pi_in,
                     const float* __restrict__ relf,
                     const f16* __restrict__ pw1, const f16* __restrict__ pw2,
                     const f16* __restrict__ pw3,
                     const float* __restrict__ b1, const float* __restrict__ b2,
                     const float* __restrict__ b3,
                     const float* __restrict__ lng, const float* __restrict__ lnb,
                     const float* __restrict__ wfin, const float* __restrict__ bfin,
                     float* __restrict__ edge_embed) {
  __shared__ f16 stage[8][16 * K1DIM];   // 48 KB: per-wave 16x192 f16 staging
  int wave = threadIdx.x >> 5;
  int lane = threadIdx.x & 31;
  int tile = blockIdx.x * 8 + wave;      // total 18432 tiles exactly
  if (tile >= BSZ * NN * (NN/16)) return;
  int jt = tile % (NN/16);
  int t2 = tile / (NN/16);
  int i  = t2 % NN;
  int b  = t2 / NN;
  f16* stg = &stage[wave][0];

  __builtin_prefetch(pw1, 0, 1);

  // stage layer-1 A: rows = 16 (i, j0+r) pairs, features [pi_i | pi_j | relfeat]
  int r    = lane >> 1;
  int hseg = lane & 1;
  int j    = jt*16 + r;
  const float* pii = pi_in + (b*NN + i)*FEAT;
  const float* pij = pi_in + (b*NN + j)*FEAT;
  const float* rfr = relf + (i - j + 383)*FEAT;
  for (int f = hseg*96; f < hseg*96 + 96; ++f) {
    float v;
    if (f < 64)       v = pii[f];
    else if (f < 128) v = pij[f - 64];
    else              v = rfr[f - 128];
    stg[r*K1DIM + f] = (f16)v;
  }
  asm volatile("s_wait_dscnt 0" ::: "memory");

  wmma_layer(stg, pw1, b1, 6, true,  lane);   // 192 -> 128, relu
  asm volatile("s_wait_dscnt 0" ::: "memory");
  wmma_layer(stg, pw2, b2, 4, true,  lane);   // 128 -> 128, relu
  asm volatile("s_wait_dscnt 0" ::: "memory");
  wmma_layer(stg, pw3, b3, 4, false, lane);   // 128 -> 128
  asm volatile("s_wait_dscnt 0" ::: "memory");

  // LayerNorm + w_fin (128->3) + sum over axis i  (2 lanes per row)
  const f16* rowp = stg + r*K1DIM + hseg*64;
  float s0 = 0.f, s1 = 0.f;
  for (int k = 0; k < 64; ++k) { float x = (float)rowp[k]; s0 += x; s1 += x*x; }
  s0 += __shfl_xor(s0, 1, 32);
  s1 += __shfl_xor(s1, 1, 32);
  float mean = s0 * (1.0f/128.0f);
  float var  = s1 * (1.0f/128.0f) - mean*mean;
  float inv  = rsqrtf(var + EPSV);
  float o0 = 0.f, o1 = 0.f, o2 = 0.f;
  for (int k = 0; k < 64; ++k) {
    int c = hseg*64 + k;
    float y = lng[c] * (((float)rowp[k]) - mean) * inv + lnb[c];
    o0 += y * wfin[c*3+0];
    o1 += y * wfin[c*3+1];
    o2 += y * wfin[c*3+2];
  }
  o0 += __shfl_xor(o0, 1, 32);
  o1 += __shfl_xor(o1, 1, 32);
  o2 += __shfl_xor(o2, 1, 32);
  if (hseg == 0) {
    float* dst = edge_embed + (b*NN + j)*3;
    atomicAdd(dst + 0, o0 + bfin[0]);
    atomicAdd(dst + 1, o1 + bfin[1]);
    atomicAdd(dst + 2, o2 + bfin[2]);
  }
}

// ------------------------------------------------- conv (SAME 3x3) + stats
__global__ void conv_cbr_kernel(const float* s0, int c0,
                                const float* s1, int c1,
                                const float* s2, int c2,
                                const float* __restrict__ kw,
                                const float* __restrict__ cb,
                                float* __restrict__ y,
                                float* __restrict__ stats) {
  int idx = blockIdx.x * blockDim.x + threadIdx.x;
  int total = BSZ * CH * HH * WW;
  if (idx >= total) return;
  int w  = idx % WW;
  int h  = (idx / WW) % HH;
  int oc = (idx / (WW*HH)) % CH;
  int b  = idx / (WW*HH*CH);
  int ict = c0 + c1 + c2;
  float acc = cb[oc];
  for (int ic = 0; ic < ict; ++ic) {
    const float* src; int icl, cx;
    if (ic < c0)            { src = s0; icl = ic;          cx = c0; }
    else if (ic < c0 + c1)  { src = s1; icl = ic - c0;     cx = c1; }
    else                    { src = s2; icl = ic - c0 - c1; cx = c2; }
    for (int kh = 0; kh < 3; ++kh) {
      int h2 = h + kh - 1;
      if (h2 < 0 || h2 >= HH) continue;
      for (int kwi = 0; kwi < 3; ++kwi) {
        int w2 = w + kwi - 1;
        if (w2 < 0 || w2 >= WW) continue;
        acc += src[((b*cx + icl)*HH + h2)*WW + w2] *
               kw[((oc*ict + ic)*3 + kh)*3 + kwi];
      }
    }
  }
  y[idx] = acc;
  atomicAdd(&stats[oc],      acc);
  atomicAdd(&stats[CH + oc], acc*acc);
}

__global__ void bn_leaky_kernel(float* __restrict__ y,
                                const float* __restrict__ stats,
                                const float* __restrict__ g,
                                const float* __restrict__ be) {
  int idx = blockIdx.x * blockDim.x + threadIdx.x;
  int total = BSZ * CH * HH * WW;
  if (idx >= total) return;
  int oc = (idx / (WW*HH)) % CH;
  const float cnt = (float)(BSZ * HH * WW);
  float mean = stats[oc] / cnt;
  float var  = stats[CH + oc] / cnt - mean*mean;
  float v = g[oc] * (y[idx] - mean) * rsqrtf(var + EPSV) + be[oc];
  y[idx] = v > 0.0f ? v : 0.01f * v;
}

// ------------------------------------------------- global pool + final linear
__global__ void final_kernel(const float* __restrict__ x5,
                             const float* __restrict__ w_l,
                             const float* __restrict__ b_l,
                             float* __restrict__ out) {
  __shared__ float pooled[CH];
  int b = blockIdx.x;
  int t = threadIdx.x;       // 32 threads, one per channel
  float s = 0.f;
  const float* base = x5 + (b*CH + t)*HH*WW;
  for (int k = 0; k < HH*WW; ++k) s += base[k];
  pooled[t] = s / (float)(HH*WW);
  __syncthreads();
  float o = b_l[t];
  for (int c = 0; c < CH; ++c) o += pooled[c] * w_l[c*CH + t];
  out[b*CH + t] = o;
}

// ---------------------------------------------------------------- launch
extern "C" void kernel_launch(void* const* d_in, const int* in_sizes, int n_in,
                              void* d_out, int out_size, void* d_ws, size_t ws_size,
                              hipStream_t stream) {
  const float* position   = (const float*)d_in[0];
  const float* charges    = (const float*)d_in[1];
  const float* atom_types = (const float*)d_in[2];
  const int*   edge_index = (const int*)  d_in[3];
  const float* w_aa   = (const float*)d_in[4];
  const float* b_aa   = (const float*)d_in[5];
  const float* w_node = (const float*)d_in[6];
  const float* b_node = (const float*)d_in[7];
  const float* w_sp   = (const float*)d_in[8];
  const float* b_sp   = (const float*)d_in[9];
  const float* w_rp   = (const float*)d_in[10];
  const float* b_rp   = (const float*)d_in[11];
  const float* w_e1   = (const float*)d_in[12];
  const float* b_e1   = (const float*)d_in[13];
  const float* w_e2   = (const float*)d_in[14];
  const float* b_e2   = (const float*)d_in[15];
  const float* w_e3   = (const float*)d_in[16];
  const float* b_e3   = (const float*)d_in[17];
  const float* ln_g   = (const float*)d_in[18];
  const float* ln_b   = (const float*)d_in[19];
  const float* w_fin  = (const float*)d_in[20];
  const float* b_fin  = (const float*)d_in[21];
  const float* k1  = (const float*)d_in[22];
  const float* cb1 = (const float*)d_in[23];
  const float* g1  = (const float*)d_in[24];
  const float* be1 = (const float*)d_in[25];
  const float* k2  = (const float*)d_in[26];
  const float* cb2 = (const float*)d_in[27];
  const float* g2  = (const float*)d_in[28];
  const float* be2 = (const float*)d_in[29];
  const float* k3  = (const float*)d_in[30];
  const float* cb3 = (const float*)d_in[31];
  const float* g3  = (const float*)d_in[32];
  const float* be3 = (const float*)d_in[33];
  const float* k4  = (const float*)d_in[34];
  const float* cb4 = (const float*)d_in[35];
  const float* g4  = (const float*)d_in[36];
  const float* be4 = (const float*)d_in[37];
  const float* k5  = (const float*)d_in[38];
  const float* cb5 = (const float*)d_in[39];
  const float* g5  = (const float*)d_in[40];
  const float* be5 = (const float*)d_in[41];
  const float* w_l = (const float*)d_in[42];
  const float* b_l = (const float*)d_in[43];
  float* out = (float*)d_out;

  // workspace carve-up (256-byte aligned slabs)
  char* base = (char*)d_ws;
  size_t cur = 0;
  auto alloc = [&](size_t bytes) {
    size_t o = cur;
    cur += (bytes + 255) & ~(size_t)255;
    return o;
  };
  float* agg        = (float*)(base + alloc(BSZ*NN*sizeof(float)));
  float* pi_buf     = (float*)(base + alloc(BSZ*NN*FEAT*sizeof(float)));
  float* relf       = (float*)(base + alloc(768*FEAT*sizeof(float)));
  float* node_embed = (float*)(base + alloc(BSZ*NN*3*sizeof(float)));
  float* edge_embed = (float*)(base + alloc(BSZ*NN*3*sizeof(float)));
  f16*   pw1        = (f16*)  (base + alloc(48*512*sizeof(f16)));
  f16*   pw2        = (f16*)  (base + alloc(32*512*sizeof(f16)));
  f16*   pw3        = (f16*)  (base + alloc(32*512*sizeof(f16)));
  float* stats      = (float*)(base + alloc(5*2*CH*sizeof(float)));
  size_t fsz = (size_t)BSZ*CH*HH*WW*sizeof(float);
  float* fw  = (float*)(base + alloc(fsz));
  float* f1  = (float*)(base + alloc(fsz));
  float* f2  = (float*)(base + alloc(fsz));
  float* f3  = (float*)(base + alloc(fsz));
  float* x5  = (float*)(base + alloc(fsz));

  // zero accumulators (every call: harness replays the graph)
  zero_f32_kernel<<<(BSZ*NN + 255)/256, 256, 0, stream>>>(agg, BSZ*NN);
  zero_f32_kernel<<<(BSZ*NN*3 + 255)/256, 256, 0, stream>>>(edge_embed, BSZ*NN*3);
  zero_f32_kernel<<<(5*2*CH + 255)/256, 256, 0, stream>>>(stats, 5*2*CH);

  // graph part
  agg_kernel<<<(NEDGE + 255)/256, 256, 0, stream>>>(position, edge_index, agg);
  node_kernel<<<BSZ*NN, 128, 0, stream>>>(charges, atom_types, w_aa, b_aa,
                                          w_node, b_node, w_sp, b_sp,
                                          agg, node_embed, pi_buf);
  relfeat_kernel<<<768, 64, 0, stream>>>(w_rp, b_rp, relf);
  pack_kernel<<<224, 256, 0, stream>>>(w_e1, w_e2, w_e3, pw1, pw2, pw3);

  // hot GEMM: 18432 wave-tiles, 8 waves / block
  edge_mlp_kernel<<<2304, 256, 0, stream>>>(pi_buf, relf, pw1, pw2, pw3,
                                            b_e1, b_e2, b_e3,
                                            ln_g, ln_b, w_fin, b_fin, edge_embed);

  // conv tower
  int cgrid = (BSZ*CH*HH*WW + 255)/256;
  conv_cbr_kernel<<<cgrid, 256, 0, stream>>>(position, 1, nullptr, 0, nullptr, 0,
                                             k1, cb1, fw, stats + 0*2*CH);
  bn_leaky_kernel<<<cgrid, 256, 0, stream>>>(fw, stats + 0*2*CH, g1, be1);

  conv_cbr_kernel<<<cgrid, 256, 0, stream>>>(fw, CH, node_embed, 1, edge_embed, 1,
                                             k2, cb2, f1, stats + 1*2*CH);
  bn_leaky_kernel<<<cgrid, 256, 0, stream>>>(f1, stats + 1*2*CH, g2, be2);

  conv_cbr_kernel<<<cgrid, 256, 0, stream>>>(fw, CH, f1, CH, nullptr, 0,
                                             k3, cb3, f2, stats + 2*2*CH);
  bn_leaky_kernel<<<cgrid, 256, 0, stream>>>(f2, stats + 2*2*CH, g3, be3);

  conv_cbr_kernel<<<cgrid, 256, 0, stream>>>(fw, CH, f1, CH, f2, CH,
                                             k4, cb4, f3, stats + 3*2*CH);
  bn_leaky_kernel<<<cgrid, 256, 0, stream>>>(f3, stats + 3*2*CH, g4, be4);

  conv_cbr_kernel<<<cgrid, 256, 0, stream>>>(f3, CH, nullptr, 0, nullptr, 0,
                                             k5, cb5, x5, stats + 4*2*CH);
  bn_leaky_kernel<<<cgrid, 256, 0, stream>>>(x5, stats + 4*2*CH, g5, be5);

  final_kernel<<<BSZ, CH, 0, stream>>>(x5, w_l, b_l, out);
}
